// CascadedGroupAttention_35218731827461
// MI455X (gfx1250) — matrix operand order repaired and checked by reading, not compiled, not run on hardware
//
#include <hip/hip_runtime.h>
#include <stdint.h>

// ---------------- problem constants ----------------
#define B_IMG   512
#define DIM     512
#define RES     14
#define NPIX    196                 // RES*RES
#define NHEADS  4
#define KD      32                  // key_dim
#define DHEAD   128                 // d
#define CIN     128                 // dim / num_heads
#define QKVO    192                 // 2*KD + DHEAD
#define NPAD    208                 // 13*16 padded query rows
#define MPAD    224                 // 14*16 padded key cols (also AV K-dim)
#define BN      (B_IMG * NPIX)      // 100352 pixels
#define SCALEQK 0.17677669529663687f // 32^-0.5

typedef __attribute__((ext_vector_type(16))) _Float16 v16h;
typedef __attribute__((ext_vector_type(8)))  float    v8f;
typedef _Float16 half_t;

// ---------------- Tensor Data Mover (CDNA5) ----------------
#if __has_builtin(__builtin_amdgcn_tensor_load_to_lds) && __has_builtin(__builtin_amdgcn_s_wait_tensorcnt)
#define USE_TDM 1
typedef __attribute__((ext_vector_type(4))) uint32_t u32x4;
typedef __attribute__((ext_vector_type(8))) int      i32x8;
typedef __attribute__((ext_vector_type(4))) int      i32x4;

// Flat pointers into LDS carry the byte offset in addr[31:0] (ISA 10.2 aperture rules).
__device__ __forceinline__ uint32_t lds_byte_off(const void* p) {
    return (uint32_t)(uintptr_t)p;
}

// D# per CDNA5 ISA: 2D tile of dwords,
// tile0_dw per row, `rows` rows, global row stride `stride_dw`, packed tight in LDS.
// rows==1 gives a plain 1D contiguous copy (tile_dim1 = 0 -> unused).
// Toolchain here exposes the 6-arg builtin: (g0 u32x4, g1 i32x8, g2 i32x4, g3 i32x4,
// unused-VADDR4 i32x8, cpol i32).
__device__ __forceinline__ void tdm_load_2d(uint32_t lds_byte, uint64_t gaddr,
                                            uint32_t tile0_dw, uint32_t rows,
                                            uint32_t stride_dw) {
    const uint32_t td0 = stride_dw;   // tensor dim0 (row length, dword units)
    const uint32_t td1 = rows;
    u32x4 g0;
    g0[0] = 1u;                                                    // count=1, user D#
    g0[1] = lds_byte;                                              // lds_addr
    g0[2] = (uint32_t)gaddr;                                       // global_addr[31:0]
    g0[3] = (uint32_t)((gaddr >> 32) & 0x01FFFFFFu) | (2u << 30);  // [56:32] | type=2
    i32x8 g1;
    g1[0] = (int)(2u << 16);                                       // data_size = 4B
    g1[1] = (int)((td0 & 0xFFFFu) << 16);                          // tensor_dim0 lo
    g1[2] = (int)(((td0 >> 16) & 0xFFFFu) | ((td1 & 0xFFFFu) << 16));
    g1[3] = (int)(((td1 >> 16) & 0xFFFFu) | (tile0_dw << 16));     // tile_dim0
    g1[4] = (int)(rows > 1u ? (rows & 0xFFFFu) : 0u);              // tile_dim1
    g1[5] = (int)stride_dw;                                        // dim0_stride lo
    g1[6] = 0;
    g1[7] = 0;
    const i32x4 gz4 = {0, 0, 0, 0};                                // <=2D: groups 2/3 zero
    const i32x8 gz8 = {0, 0, 0, 0, 0, 0, 0, 0};                    // unused VADDR4 slot
    __builtin_amdgcn_tensor_load_to_lds(g0, g1, gz4, gz4, gz8, 0);
}
#define TDM_WAIT() __builtin_amdgcn_s_wait_tensorcnt(0)
#endif

// ---------------- WMMA fragment helpers (contiguous k = 16h + s, same in A and B) ----------------
// A matrix 16x32 f16: lane l -> M = l&15, half-group h = l>>4.
// VGPR vg in 0..7 holds K = {2vg, 2vg+1} + 16h.
__device__ __forceinline__ int a_kbase(int vg, int hh) {
    int base = vg * 2;
    return base + hh * 16;
}

// Load A fragment from row-major [rows][ldk] f16 array (K contiguous).
__device__ __forceinline__ v16h load_afrag(const half_t* s, int row, int ldk, int k0, int lane) {
    union { v16h v; uint32_t u[8]; } f;
    const uint32_t* su = (const uint32_t*)s;
    const int hh = lane >> 4;
    const int base = row * ldk + k0;
#pragma unroll
    for (int vg = 0; vg < 8; ++vg)
        f.u[vg] = su[(base + a_kbase(vg, hh)) >> 1];
    return f.v;
}

// B matrix 32x16 f16: lane l -> N = l&15; halves j -> K = j + 16*(l>>4).
// Source stored N-major [ncols][ldk] with K contiguous (i.e. W[o][c]).
__device__ __forceinline__ v16h load_bfrag(const half_t* s, int ncol, int ldk, int k0, int lane) {
    union { v16h v; uint32_t u[8]; } f;
    const uint32_t* su = (const uint32_t*)s;
    const int hh = lane >> 4;
    const int base = ncol * ldk + k0 + hh * 16;
#pragma unroll
    for (int vg = 0; vg < 8; ++vg)
        f.u[vg] = su[(base + 2 * vg) >> 1];
    return f.v;
}

__device__ __forceinline__ v8f wmma_f16(v16h a, v16h b, v8f c) {
    // D = A(16x32) * B(32x16) + C, f32 accumulate
    return __builtin_amdgcn_wmma_f32_16x16x32_f16(false, a, false, b, (short)0, c, false, false);
}

// ---------------- prep kernels (BN folding, tables) ----------------
__global__ void k_fold_qkv(const float* __restrict__ w, const float* __restrict__ g,
                           const float* __restrict__ be, const float* __restrict__ mn,
                           const float* __restrict__ vr, half_t* __restrict__ wq16,
                           float* __restrict__ bq) {
    int i = blockIdx.x * blockDim.x + threadIdx.x;   // (h, o)
    if (i >= NHEADS * QKVO) return;
    float s = g[i] * rsqrtf(vr[i] + 1e-5f);
    bq[i] = be[i] - mn[i] * s;
    const float* src = w + (size_t)i * CIN;
    half_t* dst = wq16 + (size_t)i * CIN;
    for (int c = 0; c < CIN; ++c) dst[c] = (half_t)(src[c] * s);
}

__global__ void k_fold_proj(const float* __restrict__ w, const float* __restrict__ g,
                            const float* __restrict__ be, const float* __restrict__ mn,
                            const float* __restrict__ vr, half_t* __restrict__ pw16,
                            float* __restrict__ pb) {
    int o = blockIdx.x * blockDim.x + threadIdx.x;
    if (o >= DIM) return;
    float s = g[o] * rsqrtf(vr[o] + 1e-5f);
    pb[o] = be[o] - mn[o] * s;
    const float* src = w + (size_t)o * DIM;
    half_t* dst = pw16 + (size_t)o * DIM;
    for (int c = 0; c < DIM; ++c) dst[c] = (half_t)(src[c] * s);
}

__global__ void k_fold_dw(const float* __restrict__ w, const float* __restrict__ g,
                          const float* __restrict__ be, const float* __restrict__ mn,
                          const float* __restrict__ vr, float* __restrict__ dww,
                          float* __restrict__ dwb) {
    int i = blockIdx.x * blockDim.x + threadIdx.x;   // (h, c)
    if (i >= NHEADS * KD) return;
    float s = g[i] * rsqrtf(vr[i] + 1e-5f);
    dwb[i] = be[i] - mn[i] * s;
    const float* src = w + (size_t)i * 25;
    float* dst = dww + (size_t)i * 25;
    for (int j = 0; j < 25; ++j) dst[j] = src[j] * s;
}

// Expanded, padded attention-bias table: -1e30 in padded rows/cols so softmax pads vanish.
__global__ void k_build_ab(const float* __restrict__ biases, const int* __restrict__ idxs,
                           float* __restrict__ ab, int n_off) {
    int i = blockIdx.x * blockDim.x + threadIdx.x;
    if (i >= NHEADS * NPAD * MPAD) return;
    int m = i % MPAD;
    int n = (i / MPAD) % NPAD;
    int h = i / (MPAD * NPAD);
    float v = -1e30f;
    if (n < NPIX && m < NPIX) v = biases[h * n_off + idxs[n * NPIX + m]];
    ab[i] = v;
}

// Head-0 cascade input: pixel-major f16 [BN][128] from x[:, 0:128]
__global__ void k_x_to_feat(const float* __restrict__ x, half_t* __restrict__ featT) {
    int i = blockIdx.x * blockDim.x + threadIdx.x;
    if (i >= B_IMG * CIN * NPIX) return;
    int n = i % NPIX;
    int c = (i / NPIX) % CIN;
    int b = i / (NPIX * CIN);
    featT[((size_t)(b * NPIX + n)) * CIN + c] = (half_t)x[((size_t)b * DIM + c) * NPIX + n];
}

// ---------------- QKV GEMM: [BN x 128] * [128 -> 192] ----------------
// grid = BN/128 M-tiles, 256 threads = 8 waves; wave owns 16 rows x 12 column tiles.
__global__ __launch_bounds__(256) void k_qkv(const half_t* __restrict__ featT,
        const half_t* __restrict__ wq16, const float* __restrict__ bq,
        float* __restrict__ qbuf, half_t* __restrict__ kT, half_t* __restrict__ vbuf,
        int head) {
    __shared__ half_t sA[128 * 32];
    const int tid = threadIdx.x, lane = tid & 31, wave = tid >> 5;
    const int row0 = blockIdx.x * 128;
    const half_t* wqh = wq16 + (size_t)head * QKVO * CIN;
    v8f acc[12] = {};
    for (int k0 = 0; k0 < CIN; k0 += 32) {
        // stage A strip: 128 rows x 32 halves (8 KB)
#ifdef USE_TDM
        if (wave == 0)
            tdm_load_2d(lds_byte_off(sA),
                        (uint64_t)(uintptr_t)(featT + (size_t)row0 * CIN + k0),
                        /*tile0_dw=*/16, /*rows=*/128, /*stride_dw=*/CIN / 2);
        TDM_WAIT();
#else
        {
            const int u = tid * 2;
#pragma unroll
            for (int j = 0; j < 2; ++j) {
                int uu = u + j;
                int r = uu >> 2, part = uu & 3;
                const uint4* src = (const uint4*)(featT + (size_t)(row0 + r) * CIN + k0);
                ((uint4*)sA)[uu] = src[part];
            }
        }
        if (k0 + 32 < CIN)
            __builtin_prefetch(featT + (size_t)(row0 + (tid >> 1)) * CIN + k0 + 32, 0, 1);
#endif
        __syncthreads();
        v16h a = load_afrag(sA, wave * 16 + (lane & 15), 32, 0, lane);
#pragma unroll
        for (int nt = 0; nt < 12; ++nt) {
            v16h b = load_bfrag(wqh, nt * 16 + (lane & 15), CIN, k0, lane);
            acc[nt] = wmma_f16(a, b, acc[nt]);
        }
        __syncthreads();
    }
    // epilogue: split into q (f32, [b][c][n]) / k (f16, m-major) / v (f16, dd-major)
    const int pl = row0 + wave * 16;
#pragma unroll
    for (int nt = 0; nt < 12; ++nt) {
        const int o = nt * 16 + (lane & 15);
        const float bias = bq[head * QKVO + o];
#pragma unroll
        for (int r = 0; r < 8; ++r) {
            int p = pl + r + 8 * (lane >> 4);
            float val = acc[nt][r] + bias;
            int b = p / NPIX, n = p % NPIX;
            if (o < KD) {
                qbuf[((size_t)b * KD + o) * NPIX + n] = val;
            } else if (o < 2 * KD) {
                kT[((size_t)b * MPAD + n) * KD + (o - KD)] = (half_t)val;
            } else {
                vbuf[((size_t)b * DHEAD + (o - 2 * KD)) * MPAD + n] = (half_t)val;
            }
        }
    }
}

// ---------------- depthwise 5x5 conv on q (tiny) ----------------
__global__ void k_dwconv(const float* __restrict__ qbuf, const float* __restrict__ dww,
                         const float* __restrict__ dwb, half_t* __restrict__ qT, int head) {
    int i = blockIdx.x * blockDim.x + threadIdx.x;
    if (i >= B_IMG * KD * NPIX) return;
    int n = i % NPIX;
    int c = (i / NPIX) % KD;
    int b = i / (NPIX * KD);
    int y = n / RES, xx = n % RES;
    const float* wp = dww + (head * KD + c) * 25;
    const float* qp = qbuf + ((size_t)b * KD + c) * NPIX;
    float acc = dwb[head * KD + c];
#pragma unroll
    for (int dy = 0; dy < 5; ++dy) {
        int yy = y + dy - 2;
        if (yy < 0 || yy >= RES) continue;
#pragma unroll
        for (int dx = 0; dx < 5; ++dx) {
            int xc = xx + dx - 2;
            if (xc < 0 || xc >= RES) continue;
            acc += wp[dy * 5 + dx] * qp[yy * RES + xc];
        }
    }
    qT[((size_t)b * NPAD + n) * KD + c] = (half_t)acc;   // n-major for A fragments
}

// ---------------- fused attention: one workgroup per batch image ----------------
// LDS: sQ[208][32]f16 + sK[224][32]f16 + sV[128][224]f16 + sS[208][224]f32 = 271,360 B
__global__ __launch_bounds__(256) void k_attn(
        const half_t* __restrict__ qT, const half_t* __restrict__ kT,
        const half_t* __restrict__ vbuf, const float* __restrict__ ab,
        const float* __restrict__ x, half_t* __restrict__ hT,
        half_t* __restrict__ featT, int head) {
    extern __shared__ char smem[];
    half_t* sQ = (half_t*)smem;               // NPAD*KD
    half_t* sK = sQ + NPAD * KD;              // MPAD*KD
    half_t* sV = sK + MPAD * KD;              // DHEAD*MPAD
    float*  sS = (float*)(sV + DHEAD * MPAD); // NPAD*MPAD
    const int tid = threadIdx.x, lane = tid & 31, wave = tid >> 5;
    const int b = blockIdx.x;

    // stage this image's Q/K/V into LDS
#ifdef USE_TDM
    if (wave == 0)
        tdm_load_2d(lds_byte_off(sQ), (uint64_t)(uintptr_t)(qT + (size_t)b * NPAD * KD),
                    NPAD * KD / 2, 1, NPAD * KD / 2);
    else if (wave == 1)
        tdm_load_2d(lds_byte_off(sK), (uint64_t)(uintptr_t)(kT + (size_t)b * MPAD * KD),
                    MPAD * KD / 2, 1, MPAD * KD / 2);
    else if (wave == 2)
        tdm_load_2d(lds_byte_off(sV), (uint64_t)(uintptr_t)(vbuf + (size_t)b * DHEAD * MPAD),
                    DHEAD * MPAD / 2, 1, DHEAD * MPAD / 2);
    TDM_WAIT();
#else
    {
        const uint4* srcQ = (const uint4*)(qT + (size_t)b * NPAD * KD);
        uint4* dstQ = (uint4*)sQ;
        for (int u = tid; u < NPAD * KD / 8; u += 256) dstQ[u] = srcQ[u];
        const uint4* srcK = (const uint4*)(kT + (size_t)b * MPAD * KD);
        uint4* dstK = (uint4*)sK;
        for (int u = tid; u < MPAD * KD / 8; u += 256) dstK[u] = srcK[u];
        const uint4* srcV = (const uint4*)(vbuf + (size_t)b * DHEAD * MPAD);
        uint4* dstV = (uint4*)sV;
        for (int u = tid; u < DHEAD * MPAD / 8; u += 256) dstV[u] = srcV[u];
    }
#endif
    __syncthreads();

    // logits: S = (Q K) * scale + ab ; K-dim = 32 -> single wmma per 16x16 tile
    const float* abh = ab + (size_t)head * NPAD * MPAD;
    for (int t = wave; t < 13 * 14; t += 8) {
        int nt = t / 14, mt = t % 14;
        v16h aq = load_afrag(sQ, nt * 16 + (lane & 15), KD, 0, lane);
        v16h bk = load_bfrag(sK, mt * 16 + (lane & 15), KD, 0, lane);
        v8f c = {};
        c = wmma_f16(aq, bk, c);
#pragma unroll
        for (int r = 0; r < 8; ++r) {
            int n = nt * 16 + r + 8 * (lane >> 4);
            int m = mt * 16 + (lane & 15);
            sS[n * MPAD + m] = c[r] * SCALEQK + abh[n * MPAD + m];
        }
    }
    __syncthreads();

    // row softmax, then pack P as f16 in-place at the start of each f32 row
    if (tid < NPAD) {
        float* row = sS + tid * MPAD;
        float mx = -3.4e38f;
        for (int m = 0; m < MPAD; ++m) mx = fmaxf(mx, row[m]);
        float sum = 0.f;
        for (int m = 0; m < MPAD; ++m) sum += __expf(row[m] - mx);
        float inv = 1.f / sum;
        half_t* prow = (half_t*)row;           // ascending m: half store m never
        for (int m = 0; m < MPAD; ++m) {       // overlaps a future f32 load
            float e = __expf(row[m] - mx) * inv;
            prow[m] = (half_t)e;
        }
    }
    __syncthreads();

    // AV: out[dd][n] = sum_m V[dd][m] * P[n][m]; wave owns one 16-row dd tile
    const half_t* sP = (const half_t*)sS;      // row n at half offset n*(2*MPAD)
    v16h av[7];
#pragma unroll
    for (int kk = 0; kk < 7; ++kk)
        av[kk] = load_afrag(sV, wave * 16 + (lane & 15), MPAD, kk * 32, lane);
    for (int nt = 0; nt < 13; ++nt) {
        v8f c = {};
#pragma unroll
        for (int kk = 0; kk < 7; ++kk) {
            v16h bp = load_bfrag(sP, nt * 16 + (lane & 15), 2 * MPAD, kk * 32, lane);
            c = wmma_f16(av[kk], bp, c);
        }
#pragma unroll
        for (int r = 0; r < 8; ++r) {
            int dd = wave * 16 + r + 8 * (lane >> 4);
            int n = nt * 16 + (lane & 15);
            if (n < NPIX) {
                size_t p = (size_t)b * NPIX + n;
                float val = c[r];
                hT[p * DIM + head * DHEAD + dd] = (half_t)fmaxf(val, 0.f);   // ReLU'd concat
                if (head < 3) {                                              // cascade input
                    float xs = x[((size_t)b * DIM + (head + 1) * DHEAD + dd) * NPIX + n];
                    featT[p * CIN + dd] = (half_t)(val + xs);
                }
            }
        }
    }
}

// ---------------- proj GEMM: [BN x 512] * [512 -> 512] -> f32 out ----------------
__global__ __launch_bounds__(256) void k_proj(const half_t* __restrict__ hT,
        const half_t* __restrict__ pw16, const float* __restrict__ pb,
        float* __restrict__ out) {
    __shared__ half_t sA[128 * 32];
    const int tid = threadIdx.x, lane = tid & 31, wave = tid >> 5;
    const int row0 = blockIdx.x * 128;
    const int oc0 = blockIdx.y * 128;
    v8f acc[8] = {};
    for (int k0 = 0; k0 < DIM; k0 += 32) {
#ifdef USE_TDM
        if (wave == 0)
            tdm_load_2d(lds_byte_off(sA),
                        (uint64_t)(uintptr_t)(hT + (size_t)row0 * DIM + k0),
                        /*tile0_dw=*/16, /*rows=*/128, /*stride_dw=*/DIM / 2);
        TDM_WAIT();
#else
        {
            const int u = tid * 2;
#pragma unroll
            for (int j = 0; j < 2; ++j) {
                int uu = u + j;
                int r = uu >> 2, part = uu & 3;
                const uint4* src = (const uint4*)(hT + (size_t)(row0 + r) * DIM + k0);
                ((uint4*)sA)[uu] = src[part];
            }
        }
        if (k0 + 32 < DIM)
            __builtin_prefetch(hT + (size_t)(row0 + (tid >> 1)) * DIM + k0 + 32, 0, 1);
#endif
        __syncthreads();
        v16h a = load_afrag(sA, wave * 16 + (lane & 15), 32, 0, lane);
#pragma unroll
        for (int nt = 0; nt < 8; ++nt) {
            v16h bb = load_bfrag(pw16, oc0 + nt * 16 + (lane & 15), DIM, k0, lane);
            acc[nt] = wmma_f16(a, bb, acc[nt]);
        }
        __syncthreads();
    }
    const int pl = row0 + wave * 16;
#pragma unroll
    for (int nt = 0; nt < 8; ++nt) {
        const int o = oc0 + nt * 16 + (lane & 15);
        const float bias = pb[o];
#pragma unroll
        for (int r = 0; r < 8; ++r) {
            int p = pl + r + 8 * (lane >> 4);
            int b = p / NPIX, n = p % NPIX;
            out[((size_t)b * DIM + o) * NPIX + n] = acc[nt][r] + bias;
        }
    }
}

// ---------------- host ----------------
extern "C" void kernel_launch(void* const* d_in, const int* in_sizes, int n_in,
                              void* d_out, int out_size, void* d_ws, size_t ws_size,
                              hipStream_t stream) {
    (void)n_in; (void)out_size; (void)ws_size;
    const float* x      = (const float*)d_in[0];
    const float* qkv_w  = (const float*)d_in[1];
    const float* qkv_g  = (const float*)d_in[2];
    const float* qkv_b  = (const float*)d_in[3];
    const float* qkv_m  = (const float*)d_in[4];
    const float* qkv_v  = (const float*)d_in[5];
    const float* dw_w   = (const float*)d_in[6];
    const float* dw_g   = (const float*)d_in[7];
    const float* dw_b   = (const float*)d_in[8];
    const float* dw_m   = (const float*)d_in[9];
    const float* dw_v   = (const float*)d_in[10];
    const float* proj_w = (const float*)d_in[11];
    const float* proj_g = (const float*)d_in[12];
    const float* proj_b = (const float*)d_in[13];
    const float* proj_m = (const float*)d_in[14];
    const float* proj_v = (const float*)d_in[15];
    const float* att_b  = (const float*)d_in[16];
    const int*   bidx   = (const int*)d_in[17];
    const int n_off = in_sizes[16] / NHEADS;
    float* out = (float*)d_out;

    char* w = (char*)d_ws;
    size_t off = 0;
    auto alloc = [&](size_t bytes) -> char* {
        char* p = w + off;
        off += (bytes + 255) & ~(size_t)255;
        return p;
    };
    float*  ab    = (float*)alloc(sizeof(float)  * NHEADS * NPAD * MPAD);
    half_t* wq16  = (half_t*)alloc(sizeof(half_t) * NHEADS * QKVO * CIN);
    float*  bq    = (float*)alloc(sizeof(float)  * NHEADS * QKVO);
    float*  dww   = (float*)alloc(sizeof(float)  * NHEADS * KD * 25);
    float*  dwb   = (float*)alloc(sizeof(float)  * NHEADS * KD);
    half_t* pw16  = (half_t*)alloc(sizeof(half_t) * DIM * DIM);
    float*  pb    = (float*)alloc(sizeof(float)  * DIM);
    half_t* featT = (half_t*)alloc(sizeof(half_t) * (size_t)BN * CIN);
    float*  qbuf  = (float*)alloc(sizeof(float)  * (size_t)B_IMG * KD * NPIX);
    half_t* qT    = (half_t*)alloc(sizeof(half_t) * (size_t)B_IMG * NPAD * KD);
    half_t* kT    = (half_t*)alloc(sizeof(half_t) * (size_t)B_IMG * MPAD * KD);
    half_t* vbuf  = (half_t*)alloc(sizeof(half_t) * (size_t)B_IMG * DHEAD * MPAD);
    half_t* hT    = (half_t*)alloc(sizeof(half_t) * (size_t)BN * DIM);

    // zero pad regions (kT/vbuf pads feed WMMA; must be 0 so softmax/AV pads vanish)
    (void)hipMemsetAsync(qT,   0, sizeof(half_t) * (size_t)B_IMG * NPAD * KD,   stream);
    (void)hipMemsetAsync(kT,   0, sizeof(half_t) * (size_t)B_IMG * MPAD * KD,   stream);
    (void)hipMemsetAsync(vbuf, 0, sizeof(half_t) * (size_t)B_IMG * DHEAD * MPAD, stream);

    k_fold_qkv<<<(NHEADS * QKVO + 255) / 256, 256, 0, stream>>>(qkv_w, qkv_g, qkv_b, qkv_m, qkv_v, wq16, bq);
    k_fold_proj<<<(DIM + 255) / 256, 256, 0, stream>>>(proj_w, proj_g, proj_b, proj_m, proj_v, pw16, pb);
    k_fold_dw<<<1, 256, 0, stream>>>(dw_w, dw_g, dw_b, dw_m, dw_v, dww, dwb);
    k_build_ab<<<(NHEADS * NPAD * MPAD + 255) / 256, 256, 0, stream>>>(att_b, bidx, ab, n_off);
    k_x_to_feat<<<(B_IMG * CIN * NPIX + 255) / 256, 256, 0, stream>>>(x, featT);

    const size_t smem = sizeof(half_t) * (NPAD * KD + MPAD * KD + DHEAD * MPAD)
                      + sizeof(float) * NPAD * MPAD;   // 271,360 B (<320 KB/WGP)
    for (int h = 0; h < NHEADS; ++h) {
        k_qkv<<<BN / 128, 256, 0, stream>>>(featT, wq16, bq, qbuf, kT, vbuf, h);
        k_dwconv<<<(B_IMG * KD * NPIX + 255) / 256, 256, 0, stream>>>(qbuf, dww, dwb, qT, h);
        k_attn<<<B_IMG, 256, smem, stream>>>(qT, kT, vbuf, ab, x, hT, featT, h);
    }
    k_proj<<<dim3(BN / 128, DIM / 128), 256, 0, stream>>>(hT, pw16, pb, out);
}
